// PCSR_48009144435070
// MI455X (gfx1250) — matrix operand order repaired and checked by reading, not compiled, not run on hardware
//
#include <hip/hip_runtime.h>
#include <math.h>

// =====================================================================
// LIIF-style SR pipeline for MI455X (gfx1250, wave32, WMMA 16x16x32 f16)
//  - encoder res-convs: implicit-GEMM WMMA (NHWC f16 feature maps)
//  - per-pixel MLPs (heavy/light/classifier): WMMA GEMMs over 32-px tiles
//  - fused softmax-gate / select / bilinear grid-sample epilogue
// Scratch (d_ws) usage: ~5.7 MB.
// =====================================================================

typedef _Float16 f16;
typedef __attribute__((ext_vector_type(16))) _Float16 v16h;
typedef __attribute__((ext_vector_type(8)))  float    v8f;

union FragH { v16h v; unsigned u[8]; };

#define WAVES 8

// ---------------- workspace layout (bytes) ----------------
static const size_t SZ_FEAT   = (size_t)2 * 96 * 96 * 64 * 2;   // NHWC f16
static const size_t OFF_FEATA = 0;
static const size_t OFF_FEATB = OFF_FEATA + SZ_FEAT;
static const size_t OFF_RESW  = OFF_FEATB + SZ_FEAT;            // 8*9*64*64 f16
static const size_t OFF_WC1   = OFF_RESW + (size_t)8 * 9 * 64 * 64 * 2;
static const size_t OFF_WC2   = OFF_WC1 + (size_t)64 * 96 * 2;
static const size_t OFF_WL1   = OFF_WC2 + (size_t)16 * 64 * 2;
static const size_t OFF_WL2   = OFF_WL1 + (size_t)64 * 96 * 2;
static const size_t OFF_WL3   = OFF_WL2 + (size_t)64 * 64 * 2;
static const size_t OFF_WH1   = OFF_WL3 + (size_t)16 * 64 * 2;
static const size_t OFF_WM0   = OFF_WH1 + (size_t)256 * 96 * 2;
static const size_t OFF_WM1   = OFF_WM0 + (size_t)256 * 256 * 2;
static const size_t OFF_WHO   = OFF_WM1 + (size_t)256 * 256 * 2;
static const size_t OFF_CNT   = OFF_WHO + (size_t)16 * 256 * 2;

// =====================================================================
// Weight packing:  src f32 [K][M]  ->  dst f16 [Mpad][Kpad] (transposed,
// zero padded) so WMMA A-fragments read pair-contiguous dwords along K.
// =====================================================================
__global__ __launch_bounds__(256) void pack_wT_kernel(
    const float* __restrict__ src, f16* __restrict__ dst,
    int K, int M, int Kpad, int Mpad)
{
  const int i = blockIdx.x * 256 + threadIdx.x;
  if (i >= Mpad * Kpad) return;
  const int m = i / Kpad, k = i - m * Kpad;
  float v = (k < K && m < M) ? src[(size_t)k * M + m] : 0.f;
  dst[i] = (f16)v;
}

// enc_res_w f32 [8][oc=64][ic=64][3][3] -> f16 [conv][tap=ky*3+kx][oc][ic]
__global__ __launch_bounds__(256) void pack_resw_kernel(
    const float* __restrict__ src, f16* __restrict__ dst)
{
  const int d = blockIdx.x * 256 + threadIdx.x;
  if (d >= 8 * 9 * 64 * 64) return;
  const int ic   = d & 63;
  const int oc   = (d >> 6) & 63;
  const int tap  = (d >> 12) % 9;
  const int conv = d / (9 * 64 * 64);
  const int ky = tap / 3, kx = tap - ky * 3;
  const size_t si = ((((size_t)conv * 64 + oc) * 64 + ic) * 3 + ky) * 3 + kx;
  dst[d] = (f16)src[si];
}

// =====================================================================
// Input conv 3->64 (tiny), writes NHWC f16 feature map.
// =====================================================================
__global__ __launch_bounds__(256) void conv_in_kernel(
    const float* __restrict__ lr, const float* __restrict__ w,
    const float* __restrict__ bias, f16* __restrict__ feat)
{
  int i = blockIdx.x * 256 + threadIdx.x;
  if (i >= 2 * 96 * 96 * 64) return;
  const int oc = i & 63;
  int rest = i >> 6;
  const int x = rest % 96; rest /= 96;
  const int y = rest % 96; const int b = rest / 96;
  float acc = bias[oc];
  for (int ic = 0; ic < 3; ++ic)
    for (int ky = 0; ky < 3; ++ky) {
      const int iy = y + ky - 1; if (iy < 0 || iy >= 96) continue;
      for (int kx = 0; kx < 3; ++kx) {
        const int ix = x + kx - 1; if (ix < 0 || ix >= 96) continue;
        acc += lr[(((size_t)b * 3 + ic) * 96 + iy) * 96 + ix] *
               w[((oc * 3 + ic) * 3 + ky) * 3 + kx];
      }
    }
  feat[(((size_t)b * 96 + y) * 96 + x) * 64 + oc] = (f16)acc;
}

// =====================================================================
// Residual 3x3 conv 64->64 as implicit GEMM via WMMA.
// Tile: one image row, 32 pixels, all 64 oc.  8 waves -> 8 (16oc x 16px)
// tiles, 18 v_wmma per wave (9 taps x 2 ic-chunks of 32).
// LDS input tile [3 rows][34 x][64 ic] f16 with zero halo ('SAME').
// =====================================================================
__global__ __launch_bounds__(256) void resconv_kernel(
    const f16* __restrict__ src, f16* __restrict__ dst,
    const f16* __restrict__ resid, const f16* __restrict__ wPack,
    const float* __restrict__ bias, int doRelu)
{
  __shared__ __align__(16) f16 tile[3 * 34 * 64];
  const int tid = threadIdx.x;
  const int x0 = blockIdx.x * 32, y = blockIdx.y, b = blockIdx.z;

  // cooperative LDS fill (dword granularity, zero out-of-image halo)
  const unsigned* s32 = (const unsigned*)src;
  unsigned* t32 = (unsigned*)tile;
  for (int d = tid; d < 3 * 34 * 32; d += 256) {
    const int c2 = d & 31, seg = d >> 5;
    const int r = seg / 34, xx = seg - r * 34;
    const int gy = y + r - 1, gx = x0 + xx - 1;
    unsigned v = 0u;
    if (gy >= 0 && gy < 96 && gx >= 0 && gx < 96)
      v = s32[(((size_t)b * 96 + gy) * 96 + gx) * 32 + c2];
    t32[seg * 32 + c2] = v;
  }
  __syncthreads();

  const int wv = tid >> 5, lane = tid & 31;
  const int ocg = wv & 3, pxg = wv >> 2;          // 4 oc-groups x 2 px-groups
  const int mbase = ocg << 4, pxbase = pxg << 4;
  const int m = lane & 15, hi = lane >> 4;        // m doubles as B-col n

  v8f acc = {0.f, 0.f, 0.f, 0.f, 0.f, 0.f, 0.f, 0.f};
  for (int ky = 0; ky < 3; ++ky)
    for (int kx = 0; kx < 3; ++kx) {
      const unsigned* wrow =
          (const unsigned*)(wPack + (size_t)((ky * 3 + kx) * 64 + mbase + m) * 64);
      for (int kc = 0; kc < 64; kc += 32) {
        FragH a, bf;
        const int g0 = (kc + hi * 8) >> 1;
        const int g1 = (kc + 16 + hi * 8) >> 1;
#pragma unroll
        for (int i = 0; i < 4; ++i) { a.u[i] = wrow[g0 + i]; a.u[4 + i] = wrow[g1 + i]; }
        const unsigned* bp = (const unsigned*)tile +
            ((((ky * 34) + (pxbase + m + kx)) * 64 + kc + hi * 8) >> 1);
#pragma unroll
        for (int i = 0; i < 4; ++i) { bf.u[i] = bp[i]; bf.u[4 + i] = bp[8 + i]; }
        acc = __builtin_amdgcn_wmma_f32_16x16x32_f16(
            false, a.v, false, bf.v, (short)0, acc, false, false);
      }
    }

  const int px = x0 + pxbase + m;
#pragma unroll
  for (int r = 0; r < 8; ++r) {
    const int oc = mbase + r + hi * 8;
    float v = acc[r] + bias[oc];
    const size_t oi = (((size_t)b * 96 + y) * 96 + px) * 64 + oc;
    if (resid) v += (float)resid[oi];
    if (doRelu) v = fmaxf(v, 0.f);
    dst[oi] = (f16)v;
  }
}

// =====================================================================
// Generic LDS-activation GEMM stage:  out[M, 32px] = W[M,K] * act[K,32px]
// act LDS layout [px][K] (pair-contiguous along K for B-fragments).
// =====================================================================
__device__ __forceinline__ void mlp_gemm(
    const f16* actIn, int K, const f16* __restrict__ wT,
    const float* __restrict__ bias, int Mreal,
    f16* actOut, int Mpad, bool relu, int wv, int lane)
{
  const int n16 = lane & 15, hi = lane >> 4;
  const int nTiles = (Mpad >> 4) * 2;            // 2 pixel groups of 16
  for (int t = wv; t < nTiles; t += WAVES) {     // wave-uniform loop
    const int mg = t >> 1, ng = t & 1;
    const int mbase = mg << 4, pxbase = ng << 4;
    const unsigned* wrow = (const unsigned*)(wT + (size_t)(mbase + n16) * K);
    const unsigned* brow = (const unsigned*)(actIn + (size_t)(pxbase + n16) * K);
    v8f acc = {0.f, 0.f, 0.f, 0.f, 0.f, 0.f, 0.f, 0.f};
    for (int kb = 0; kb < K; kb += 32) {
      FragH a, bf;
      const int g0 = (kb + hi * 8) >> 1;
      const int g1 = (kb + 16 + hi * 8) >> 1;
#pragma unroll
      for (int i = 0; i < 4; ++i) { a.u[i] = wrow[g0 + i]; a.u[4 + i] = wrow[g1 + i]; }
      const int bb = (kb + hi * 8) >> 1;
#pragma unroll
      for (int i = 0; i < 4; ++i) { bf.u[i] = brow[bb + i]; bf.u[4 + i] = brow[bb + 8 + i]; }
      acc = __builtin_amdgcn_wmma_f32_16x16x32_f16(
          false, a.v, false, bf.v, (short)0, acc, false, false);
    }
#pragma unroll
    for (int r = 0; r < 8; ++r) {
      const int oc = mbase + r + hi * 8;
      float v = acc[r] + (oc < Mreal ? bias[oc] : 0.f);
      if (relu) v = fmaxf(v, 0.f);
      actOut[(size_t)(pxbase + n16) * Mpad + oc] = (f16)v;
    }
  }
}

// =====================================================================
// Fused make_inp + classifier + light + heavy MLPs + select + bilinear
// grid-sample.  One block = 32 consecutive output pixels of one row.
// =====================================================================
__global__ __launch_bounds__(256) void mlp_kernel(
    const f16* __restrict__ feat, const float* __restrict__ coord,
    const float* __restrict__ cell, const float* __restrict__ lr,
    const f16* wc1, const float* bc1, const f16* wc2, const float* bc2,
    const f16* wl1, const float* bl1, const f16* wl2, const float* bl2,
    const f16* wl3, const float* bl3,
    const f16* wh1, const float* bh1, const f16* wm0, const f16* wm1,
    const float* bmid, const f16* who, const float* bho,
    float* __restrict__ out, unsigned* __restrict__ gCnt)
{
  __shared__ __align__(16) f16 sInp[32 * 96];
  __shared__ __align__(16) f16 sA[32 * 256];
  __shared__ __align__(16) f16 sB[32 * 256];
  __shared__ __align__(16) f16 sC[32 * 64];
  __shared__ __align__(16) f16 sOH[32 * 16];
  __shared__ __align__(16) f16 sOL[32 * 16];
  __shared__ __align__(16) f16 sOC[32 * 16];
  __shared__ unsigned sCnt;

  const int tid = threadIdx.x;
  const int b = blockIdx.z, j = blockIdx.y, x0 = blockIdx.x * 32;
  if (tid == 0) sCnt = 0;

  // ---- build padded inp tile [32 px][96]: 64 feat + rel(2) + cell(2) + 0s
  {
    const int px = tid >> 3, part = tid & 7;
    const int gx = x0 + px;
    const size_t ci = (((size_t)b * 384 + j) * 384 + gx) * 2;
    const float c0 = coord[ci], c1 = coord[ci + 1];
    const float yl = (c0 + 1.f) * 48.f - 0.5f;
    const float xl = (c1 + 1.f) * 48.f - 0.5f;
    int iyn = (int)floorf(yl + 0.5f); iyn = min(max(iyn, 0), 95);
    int ixn = (int)floorf(xl + 0.5f); ixn = min(max(ixn, 0), 95);
    const unsigned* fsrc =
        (const unsigned*)feat + (((size_t)b * 96 + iyn) * 96 + ixn) * 32;
    unsigned* idst = (unsigned*)sInp + px * 48;
    for (int d = part * 6; d < part * 6 + 6; ++d) {
      if (d < 32) {
        idst[d] = fsrc[d];
      } else if (d == 32) {
        const float cyc = (2.f * iyn + 1.f) / 96.f - 1.f;
        const float cxc = (2.f * ixn + 1.f) / 96.f - 1.f;
        sInp[px * 96 + 64] = (f16)((c0 - cyc) * 96.f);
        sInp[px * 96 + 65] = (f16)((c1 - cxc) * 96.f);
      } else if (d == 33) {
        sInp[px * 96 + 66] = (f16)(cell[ci] * 96.f);
        sInp[px * 96 + 67] = (f16)(cell[ci + 1] * 96.f);
      } else {
        idst[d] = 0u;
      }
    }
  }
  __syncthreads();

  const int wv = tid >> 5, lane = tid & 31;
  // heavy path
  mlp_gemm(sInp, 96,  wh1, bh1,        256, sA,  256, true,  wv, lane); __syncthreads();
  mlp_gemm(sA,  256,  wm0, bmid,       256, sB,  256, true,  wv, lane); __syncthreads();
  mlp_gemm(sB,  256,  wm1, bmid + 256, 256, sA,  256, true,  wv, lane); __syncthreads();
  mlp_gemm(sA,  256,  who, bho,          3, sOH,  16, false, wv, lane); __syncthreads();
  // classifier
  mlp_gemm(sInp, 96,  wc1, bc1,         64, sC,   64, true,  wv, lane); __syncthreads();
  mlp_gemm(sC,   64,  wc2, bc2,          2, sOC,  16, false, wv, lane); __syncthreads();
  // light path (reuse sB low region as [32][64])
  mlp_gemm(sInp, 96,  wl1, bl1,         64, sB,   64, true,  wv, lane); __syncthreads();
  mlp_gemm(sB,   64,  wl2, bl2,         64, sC,   64, true,  wv, lane); __syncthreads();
  mlp_gemm(sC,   64,  wl3, bl3,          3, sOL,  16, false, wv, lane); __syncthreads();

  // ---- epilogue: gate, select, bilinear grid-sample, store
  if (tid < 32) {
    const int px = tid, gx = x0 + px;
    const float l0 = (float)sOC[px * 16 + 0], l1 = (float)sOC[px * 16 + 1];
    const bool hard = l1 > l0;                 // softmax(l)[1] > 0.5
    if (!hard) atomicAdd(&sCnt, 1u);
    const size_t ci = (((size_t)b * 384 + j) * 384 + gx) * 2;
    const float gyn = coord[ci], gxn = coord[ci + 1];   // flip(-1) -> (x,y)
    const float xs = (gxn + 1.f) * 48.f - 0.5f;
    const float ys = (gyn + 1.f) * 48.f - 0.5f;
    const float xf = floorf(xs), yf = floorf(ys);
    const float wx = xs - xf, wy = ys - yf;
    const int x0i = min(max((int)xf, 0), 95), x1i = min(max((int)xf + 1, 0), 95);
    const int y0i = min(max((int)yf, 0), 95), y1i = min(max((int)yf + 1, 0), 95);
#pragma unroll
    for (int c = 0; c < 3; ++c) {
      const float* ch = lr + ((size_t)b * 3 + c) * 96 * 96;
      const float v00 = ch[y0i * 96 + x0i], v01 = ch[y0i * 96 + x1i];
      const float v10 = ch[y1i * 96 + x0i], v11 = ch[y1i * 96 + x1i];
      const float bil = v00 * (1.f - wx) * (1.f - wy) + v01 * wx * (1.f - wy) +
                        v10 * (1.f - wx) * wy + v11 * wx * wy;
      const float pv = hard ? (float)sOH[px * 16 + c] : (float)sOL[px * 16 + c];
      out[(((size_t)b * 3 + c) * 384 + j) * 384 + gx] = pv + bil;
    }
  }
  __syncthreads();
  if (tid == 0) atomicAdd(gCnt, sCnt);
}

__global__ void zero_cnt_kernel(unsigned* c) { *c = 0u; }
__global__ void finalize_kernel(float* ratio, const unsigned* c) {
  *ratio = (float)(*c) * (1.0f / 294912.0f);
}

// =====================================================================
extern "C" void kernel_launch(void* const* d_in, const int* in_sizes, int n_in,
                              void* d_out, int out_size, void* d_ws, size_t ws_size,
                              hipStream_t stream) {
  (void)in_sizes; (void)n_in; (void)out_size; (void)ws_size;
  char* ws = (char*)d_ws;
  const float* lr       = (const float*)d_in[0];
  const float* coord    = (const float*)d_in[1];
  const float* cellp    = (const float*)d_in[2];
  const float* w_in     = (const float*)d_in[3];
  const float* b_in     = (const float*)d_in[4];
  const float* res_w    = (const float*)d_in[5];
  const float* res_b    = (const float*)d_in[6];
  float* out = (float*)d_out;

  f16* featA = (f16*)(ws + OFF_FEATA);
  f16* featB = (f16*)(ws + OFF_FEATB);
  f16* resw  = (f16*)(ws + OFF_RESW);
  unsigned* cnt = (unsigned*)(ws + OFF_CNT);

  // ---- pack weights to f16 (transposed, K-padded) ----
  pack_resw_kernel<<<(8 * 9 * 64 * 64 + 255) / 256, 256, 0, stream>>>(res_w, resw);
  auto packW = [&](const float* src, size_t off, int K, int M, int Kp, int Mp) {
    const int tot = Mp * Kp;
    pack_wT_kernel<<<(tot + 255) / 256, 256, 0, stream>>>(src, (f16*)(ws + off), K, M, Kp, Mp);
  };
  packW((const float*)d_in[7],  OFF_WC1,  68,  64,  96,  64);   // cls_w1
  packW((const float*)d_in[9],  OFF_WC2,  64,   2,  64,  16);   // cls_w2
  packW((const float*)d_in[11], OFF_WL1,  68,  64,  96,  64);   // lw1
  packW((const float*)d_in[13], OFF_WL2,  64,  64,  64,  64);   // lw2
  packW((const float*)d_in[15], OFF_WL3,  64,   3,  64,  16);   // lw3
  packW((const float*)d_in[17], OFF_WH1,  68, 256,  96, 256);   // hw1
  packW((const float*)d_in[19],             OFF_WM0, 256, 256, 256, 256);
  packW((const float*)d_in[19] + 256 * 256, OFF_WM1, 256, 256, 256, 256);
  packW((const float*)d_in[21], OFF_WHO, 256,   3, 256,  16);   // hw_out

  // ---- encoder ----
  conv_in_kernel<<<(2 * 96 * 96 * 64) / 256, 256, 0, stream>>>(lr, w_in, b_in, featA);
  const dim3 cgrid(3, 96, 2);
  for (int blk = 0; blk < 4; ++blk) {
    resconv_kernel<<<cgrid, 256, 0, stream>>>(
        featA, featB, nullptr, resw + (size_t)(2 * blk) * 9 * 64 * 64,
        res_b + (size_t)(2 * blk) * 64, 1);
    resconv_kernel<<<cgrid, 256, 0, stream>>>(
        featB, featA, featA, resw + (size_t)(2 * blk + 1) * 9 * 64 * 64,
        res_b + (size_t)(2 * blk + 1) * 64, 0);
  }

  // ---- fused MLP + epilogue ----
  zero_cnt_kernel<<<1, 1, 0, stream>>>(cnt);
  mlp_kernel<<<dim3(12, 384, 2), 256, 0, stream>>>(
      featA, coord, cellp, lr,
      (const f16*)(ws + OFF_WC1), (const float*)d_in[8],
      (const f16*)(ws + OFF_WC2), (const float*)d_in[10],
      (const f16*)(ws + OFF_WL1), (const float*)d_in[12],
      (const f16*)(ws + OFF_WL2), (const float*)d_in[14],
      (const f16*)(ws + OFF_WL3), (const float*)d_in[16],
      (const f16*)(ws + OFF_WH1), (const float*)d_in[18],
      (const f16*)(ws + OFF_WM0), (const f16*)(ws + OFF_WM1),
      (const float*)d_in[20],
      (const f16*)(ws + OFF_WHO), (const float*)d_in[22],
      out, cnt);
  finalize_kernel<<<1, 1, 0, stream>>>(out + (size_t)2 * 3 * 384 * 384, cnt);
}
